// NASH_18605798326720
// MI455X (gfx1250) — matrix operand order repaired and checked by reading, not compiled, not run on hardware
//
#include <hip/hip_runtime.h>
#include <math.h>

// Problem sizes (match reference)
#define BB 512
#define VV 50000
#define HH 500
#define DD 128
#define NP 512      // H padded to multiple of tile N
#define SPLITS 8
#define KSPAN 6272  // 196*32: multiple of TK so main loops are guard-free

// GEMM1 tiling
#define TM 128
#define TN 64
#define TK 32
#define SA 36       // LDS row stride (shorts), even -> dword-aligned pair reads
#define SB 36

typedef __attribute__((ext_vector_type(16))) __bf16 v16bf;
typedef __attribute__((ext_vector_type(8)))  float  v8f;

// Native RNE conversions -> backend can emit v_cvt_pk_bf16_f32 / v_fma_mix
__device__ __forceinline__ unsigned short bf16_rne(float f) {
  __bf16 h = (__bf16)f;
  return __builtin_bit_cast(unsigned short, h);
}
__device__ __forceinline__ float bf16_f32(unsigned short u) {
  return (float)__builtin_bit_cast(__bf16, u);
}

// -------------------------------------------------------------------------
// Kernel 1: n1[b] = count of nonzeros in x row b (collapsed-softmax statistic)
// -------------------------------------------------------------------------
__global__ __launch_bounds__(256) void row_count(const float* __restrict__ x,
                                                 float* __restrict__ n1) {
  __shared__ float red[256];
  const int b = blockIdx.x;
  float c = 0.0f;
  for (int i = threadIdx.x; i < VV; i += 256)
    c += (x[(size_t)b * VV + i] != 0.0f) ? 1.0f : 0.0f;
  red[threadIdx.x] = c; __syncthreads();
  for (int s = 128; s > 0; s >>= 1) {
    if (threadIdx.x < s) red[threadIdx.x] += red[threadIdx.x + s];
    __syncthreads();
  }
  if (threadIdx.x == 0) n1[b] = red[0];
}

// -------------------------------------------------------------------------
// Kernel 2: split-K WMMA GEMM  hpart[s] = x[:, ks:ke] @ W1[ks:ke, :]
//   x exact in bf16; W1 split into bf16 hi+lo -> near-fp32 accuracy with f32
//   WMMA accumulation. 8 waves, 128x64 tile, double-buffered LDS staging,
//   vectorized (b128) global loads, guard-free main loop (tail only in last
//   split). Each wave: 16x64 strip = 4 C tiles, 8 v_wmma per K-step.
// -------------------------------------------------------------------------
__global__ __launch_bounds__(256) void gemm1_wmma(const float* __restrict__ x,
                                                  const float* __restrict__ W1,
                                                  float* __restrict__ hpart) {
  __shared__ __align__(16) unsigned short lsA [2][TM * SA];  // x tile, bf16 (exact)
  __shared__ __align__(16) unsigned short lsBh[2][TN * SB];  // W1 hi, transposed [n][k]
  __shared__ __align__(16) unsigned short lsBl[2][TN * SB];  // W1 lo, transposed [n][k]

  const int tid  = threadIdx.x;
  const int lane = tid & 31;
  const int wave = tid >> 5;
  const int m0   = blockIdx.y * TM;
  const int n0   = blockIdx.x * TN;
  const int s    = blockIdx.z;
  const int kBeg = s * KSPAN;
  const int kEnd = (kBeg + KSPAN < VV) ? (kBeg + KSPAN) : VV;

  v8f acc[4];
  #pragma unroll
  for (int j = 0; j < 4; ++j)
    #pragma unroll
    for (int e = 0; e < 8; ++e) acc[j][e] = 0.0f;

  // ---- global fetch (registers), guard-free, 16B vector loads ----
  auto fetchX = [&](int k0, float4* xv) {
    #pragma unroll
    for (int i = 0; i < 4; ++i) {
      int idx = i * 256 + tid;
      int r = idx >> 3, c4 = (idx & 7) << 2;                 // 8 threads/row
      xv[i] = *(const float4*)&x[(size_t)(m0 + r) * VV + (k0 + c4)];
    }
  };
  auto fetchW = [&](int k0, float4* wv) {
    const size_t lim = (size_t)VV * HH - 4;
    #pragma unroll
    for (int i = 0; i < 2; ++i) {
      int idx = i * 256 + tid;
      int r = idx >> 4, c4 = (idx & 15) << 2;                // 16 threads/row
      size_t gi = (size_t)(k0 + r) * HH + (size_t)(n0 + c4);
      if (gi > lim) gi = lim;  // branchless N-edge clamp: only padded cols clamp
      wv[i] = *(const float4*)&W1[gi];
    }
  };
  // ---- convert + stage into LDS ----
  auto stageX = [&](int buf, const float4* xv) {
    #pragma unroll
    for (int i = 0; i < 4; ++i) {
      int idx = i * 256 + tid;
      int r = idx >> 3, c4 = (idx & 7) << 2;
      unsigned int p0 = (unsigned int)bf16_rne(xv[i].x) | ((unsigned int)bf16_rne(xv[i].y) << 16);
      unsigned int p1 = (unsigned int)bf16_rne(xv[i].z) | ((unsigned int)bf16_rne(xv[i].w) << 16);
      *(uint2*)&lsA[buf][r * SA + c4] = make_uint2(p0, p1);  // 8B-aligned: 72|8
    }
  };
  auto stageW = [&](int buf, const float4* wv) {
    #pragma unroll
    for (int i = 0; i < 2; ++i) {
      int idx = i * 256 + tid;
      int r = idx >> 4, c4 = (idx & 15) << 2;
      const float w4[4] = {wv[i].x, wv[i].y, wv[i].z, wv[i].w};
      #pragma unroll
      for (int j = 0; j < 4; ++j) {
        unsigned short hi = bf16_rne(w4[j]);
        lsBh[buf][(c4 + j) * SB + r] = hi;                   // transposed store
        lsBl[buf][(c4 + j) * SB + r] = bf16_rne(w4[j] - bf16_f32(hi));
      }
    }
  };
  // ---- WMMA on one staged tile ----
  auto computeTile = [&](int buf) {
    union U { unsigned int u[8]; v16bf v; };
    U fa;
    const int m = wave * 16 + (lane & 15);
    const int g = lane >> 4;
    #pragma unroll
    for (int r = 0; r < 8; ++r) {       // A 16x32 bf16 layout: K-pairs per VGPR
      int kk = ((r >> 2) << 4) | (g << 3) | ((r & 3) << 1);
      fa.u[r] = *(const unsigned int*)&lsA[buf][m * SA + kk];
    }
    #pragma unroll
    for (int j = 0; j < 4; ++j) {
      U fbh, fbl;
      const int n = j * 16 + (lane & 15);
      #pragma unroll
      for (int r = 0; r < 8; ++r) {     // B 32x16: col=lane%16, K-pairs per VGPR (same order as A)
        int kk = ((r >> 2) << 4) | (g << 3) | ((r & 3) << 1);
        fbh.u[r] = *(const unsigned int*)&lsBh[buf][n * SB + kk];
        fbl.u[r] = *(const unsigned int*)&lsBl[buf][n * SB + kk];
      }
      acc[j] = __builtin_amdgcn_wmma_f32_16x16x32_bf16(
          false, fa.v, false, fbh.v, (short)0, acc[j], false, false);
      acc[j] = __builtin_amdgcn_wmma_f32_16x16x32_bf16(
          false, fa.v, false, fbl.v, (short)0, acc[j], false, false);
    }
  };

  const int nFull = (kEnd - kBeg) / TK;
  const int kTail = kBeg + nFull * TK;

  float4 xv[4], wv[2];
  int cur = 0;
  fetchX(kBeg, xv); fetchW(kBeg, wv);
  stageX(0, xv);    stageW(0, wv);

  for (int st = 0; st < nFull; ++st) {
    const bool pf = (st + 1 < nFull);
    if (pf) { fetchX(kBeg + (st + 1) * TK, xv); fetchW(kBeg + (st + 1) * TK, wv); }
    __syncthreads();                    // buffer `cur` ready for all waves
    computeTile(cur);
    if (pf) { stageX(cur ^ 1, xv); stageW(cur ^ 1, wv); }
    cur ^= 1;
  }

  if (kTail < kEnd) {                   // only last split: 16 leftover K, guarded
    #pragma unroll
    for (int i = 0; i < (TM * TK) / 256; ++i) {
      int idx = i * 256 + tid;
      int r = idx >> 5, c = idx & 31;
      int gk = kTail + c;
      float v = (gk < kEnd) ? x[(size_t)(m0 + r) * VV + gk] : 0.0f;
      lsA[cur][r * SA + c] = bf16_rne(v);
    }
    #pragma unroll
    for (int i = 0; i < (TK * TN) / 256; ++i) {
      int idx = i * 256 + tid;
      int r = idx >> 6, c = idx & 63;
      int gk = kTail + r, gn = n0 + c;
      float w = (gk < kEnd && gn < HH) ? W1[(size_t)gk * HH + gn] : 0.0f;
      unsigned short hi = bf16_rne(w);
      lsBh[cur][c * SB + r] = hi;
      lsBl[cur][c * SB + r] = bf16_rne(w - bf16_f32(hi));
    }
    __syncthreads();
    computeTile(cur);
  }

  // ---- store partials: C layout row = r + 8*(lane/16), col = lane%16 ----
  {
    const int g  = lane >> 4;
    const int nl = lane & 15;
    #pragma unroll
    for (int j = 0; j < 4; ++j)
      #pragma unroll
      for (int r = 0; r < 8; ++r) {
        int m = m0 + wave * 16 + r + 8 * g;
        int n = n0 + j * 16 + nl;
        hpart[((size_t)s * BB + m) * NP + n] = acc[j][r];
      }
  }
}

// -------------------------------------------------------------------------
// Kernel 3: h = relu(sum_s hpart[s] + b1)   (split-K reduce, deterministic)
// -------------------------------------------------------------------------
__global__ __launch_bounds__(256) void bias_relu_reduce(const float* __restrict__ hpart,
                                                        const float* __restrict__ b1,
                                                        float* __restrict__ h) {
  int idx = blockIdx.x * 256 + threadIdx.x;      // over BB*NP
  int b = idx >> 9, j = idx & (NP - 1);
  float v = 0.0f;
  if (j < HH) {
    v = b1[j];
    #pragma unroll
    for (int s = 0; s < SPLITS; ++s) v += hpart[((size_t)s * BB + b) * NP + j];
    v = fmaxf(v, 0.0f);
  }
  h[idx] = v;
}

// -------------------------------------------------------------------------
// Kernel 4: z = sigmoid(h@W2 + b2), code = step(z-0.5), KL row partials.
//   Tiny (3.3e7 MAC) and precision-sensitive -> exact fp32 VALU.
// -------------------------------------------------------------------------
__global__ __launch_bounds__(DD) void encode2(const float* __restrict__ h,
                                              const float* __restrict__ W2,
                                              const float* __restrict__ b2,
                                              float* __restrict__ z,
                                              float* __restrict__ code,
                                              float* __restrict__ klp) {
  __shared__ float hrow[HH];
  __shared__ float red[DD];
  const int b = blockIdx.x, d = threadIdx.x;
  for (int i = d; i < HH; i += DD) hrow[i] = h[(size_t)b * NP + i];
  __syncthreads();
  float acc = b2[d];
  for (int k = 0; k < HH; ++k) acc = fmaf(hrow[k], W2[k * DD + d], acc);
  float zz = 1.0f / (1.0f + expf(-acc));
  float cd = (zz > 0.5f) ? 1.0f : ((zz < 0.5f) ? 0.0f : 0.5f);  // sign(0)->0.5
  z[b * DD + d] = zz;
  code[b * DD + d] = cd;
  // z*log(z/0.5) + (1-z)*log((1-z)/0.5)
  red[d] = zz * logf(2.0f * zz) + (1.0f - zz) * logf(2.0f * (1.0f - zz));
  __syncthreads();
  for (int s = DD / 2; s > 0; s >>= 1) {
    if (d < s) red[d] += red[d + s];
    __syncthreads();
  }
  if (d == 0) klp[b] = red[0];
}

// -------------------------------------------------------------------------
// Kernel 5: z_noise = code + z@Wstd + bstd; d0 = <z_noise,E[0]>, d1 = <.,E[1]>
// -------------------------------------------------------------------------
__global__ __launch_bounds__(DD) void noise_dots(const float* __restrict__ z,
                                                 const float* __restrict__ code,
                                                 const float* __restrict__ Wstd,
                                                 const float* __restrict__ bstd,
                                                 const float* __restrict__ E,
                                                 float* __restrict__ d0,
                                                 float* __restrict__ d1) {
  __shared__ float zrow[DD], znrow[DD], red[DD];
  const int b = blockIdx.x, d = threadIdx.x;
  zrow[d] = z[b * DD + d];
  __syncthreads();
  float acc = bstd[d];
  for (int k = 0; k < DD; ++k) acc = fmaf(zrow[k], Wstd[k * DD + d], acc);
  float zn = code[b * DD + d] + acc;
  znrow[d] = zn;
  red[d] = zn * E[d];                 // E row 0
  __syncthreads();
  for (int s = DD / 2; s > 0; s >>= 1) {
    if (d < s) red[d] += red[d + s];
    __syncthreads();
  }
  if (d == 0) d0[b] = red[0];
  __syncthreads();
  red[d] = znrow[d] * E[DD + d];      // E row 1
  __syncthreads();
  for (int s = DD / 2; s > 0; s >>= 1) {
    if (d < s) red[d] += red[d + s];
    __syncthreads();
  }
  if (d == 0) d1[b] = red[0];
}

// -------------------------------------------------------------------------
// Kernel 6: analytic two-value log-softmax loss + KL total.
//   lse = m + log((V-n1)e^{d0-m} + n1 e^{d1-m}); loss = -(1/B) sum n1(d1-lse)
// -------------------------------------------------------------------------
__global__ __launch_bounds__(BB) void finalize(const float* __restrict__ d0,
                                               const float* __restrict__ d1,
                                               const float* __restrict__ n1,
                                               const float* __restrict__ klp,
                                               float* __restrict__ out) {
  __shared__ float rl[BB], rk[BB];
  const int t = threadIdx.x;
  float nn = n1[t];
  float a = d0[t], b = d1[t];
  float m = fmaxf(a, b);
  float lse = m + logf(((float)VV - nn) * expf(a - m) + nn * expf(b - m));
  rl[t] = nn * (b - lse);
  rk[t] = klp[t];
  __syncthreads();
  for (int s = BB / 2; s > 0; s >>= 1) {
    if (t < s) { rl[t] += rl[t + s]; rk[t] += rk[t + s]; }
    __syncthreads();
  }
  if (t == 0) { out[0] = -rl[0] / (float)BB; out[1] = rk[0]; }
}

// -------------------------------------------------------------------------
extern "C" void kernel_launch(void* const* d_in, const int* in_sizes, int n_in,
                              void* d_out, int out_size, void* d_ws, size_t ws_size,
                              hipStream_t stream) {
  const float* x    = (const float*)d_in[0];
  const float* W1   = (const float*)d_in[1];
  const float* b1   = (const float*)d_in[2];
  const float* W2   = (const float*)d_in[3];
  const float* b2   = (const float*)d_in[4];
  const float* E    = (const float*)d_in[5];
  const float* Wstd = (const float*)d_in[6];
  const float* bstd = (const float*)d_in[7];
  float* out = (float*)d_out;

  float* ws = (float*)d_ws;
  size_t off = 0;
  float* hpart = ws + off; off += (size_t)SPLITS * BB * NP;  // 8 MB
  float* h     = ws + off; off += (size_t)BB * NP;           // 1 MB
  float* z     = ws + off; off += (size_t)BB * DD;
  float* code  = ws + off; off += (size_t)BB * DD;
  float* klp   = ws + off; off += (size_t)BB;
  float* d0    = ws + off; off += (size_t)BB;
  float* d1    = ws + off; off += (size_t)BB;
  float* n1    = ws + off; off += (size_t)BB;

  row_count<<<BB, 256, 0, stream>>>(x, n1);
  gemm1_wmma<<<dim3(NP / TN, BB / TM, SPLITS), 256, 0, stream>>>(x, W1, hpart);
  bias_relu_reduce<<<(BB * NP) / 256, 256, 0, stream>>>(hpart, b1, h);
  encode2<<<BB, DD, 0, stream>>>(h, W2, b2, z, code, klp);
  noise_dots<<<BB, DD, 0, stream>>>(z, code, Wstd, bstd, E, d0, d1);
  finalize<<<1, BB, 0, stream>>>(d0, d1, n1, klp, out);
}
